// MinGRUBlock_77309411814
// MI455X (gfx1250) — matrix operand order repaired and checked by reading, not compiled, not run on hardware
//
#include <hip/hip_runtime.h>
#include <hip/hip_bf16.h>
#include <cstdint>

// ---------------------------------------------------------------------------
// MinGRU block on gfx1250 (MI455X):
//   - all GEMMs via v_wmma_f32_16x16x32_bf16 (fp32 accumulate)
//   - global->LDS staging via async direct path (global_load_async_to_lds_b128,
//     ASYNCcnt) with double-buffered tiles overlapping WMMA compute
// ---------------------------------------------------------------------------

typedef __attribute__((ext_vector_type(16))) __bf16 v16bf;
typedef __attribute__((ext_vector_type(8)))  __bf16 v8bf;
typedef __attribute__((ext_vector_type(8)))  float  v8f;

#define DM    1024
#define DFF   4096
#define NB    4
#define LSEQ  4096
#define MTOK  (NB * LSEQ)     // 16384 tokens
#define EPSV  1e-6f

#define BM 128
#define BN 64
#define BK 32
#define LDS_A (BK + 8)        // padded stride (bf16 elems) -> conflict-free
#define LDS_B (BK + 8)

union FragU { v16bf v; v8bf h[2]; };

__device__ __forceinline__ float fsigmoid(float x) { return 1.0f / (1.0f + __expf(-x)); }

// --- CDNA5 async global->LDS direct copy (16B), tracked with ASYNCcnt -------
__device__ __forceinline__ void async_copy16(const __bf16* gsrc, __bf16* ldst) {
    const uint32_t lds_off = (uint32_t)(uintptr_t)ldst;   // low 32 bits = LDS offset
    const uint64_t gaddr   = (uint64_t)(uintptr_t)gsrc;
    asm volatile("global_load_async_to_lds_b128 %0, %1, off"
                 :: "v"(lds_off), "v"(gaddr) : "memory");
}
__device__ __forceinline__ void wait_async0() {
    asm volatile("s_wait_asynccnt 0x0" ::: "memory");
}

// Load A fragment (16x32 bf16, contiguous k = 16*(lane/16) + s, same as B) from padded LDS tile.
__device__ __forceinline__ v16bf load_frag_a(const __bf16* sm, int rowBase, int lane) {
    const int lm = lane & 15;
    const int kb = (lane & 16) ? 16 : 0;
    const int r  = (rowBase + lm) * LDS_A;
    FragU u;
    u.h[0] = *(const v8bf*)(sm + r + kb);        // K = kb .. kb+7
    u.h[1] = *(const v8bf*)(sm + r + kb + 8);    // K = kb+8 .. kb+15
    return u.v;
}

// Load B fragment (32x16 bf16) from padded LDS tile holding B^T rows (N x K).
__device__ __forceinline__ v16bf load_frag_b(const __bf16* sm, int colBase, int lane) {
    const int lm = lane & 15;
    const int kb = (lane & 16) ? 16 : 0;
    const int r  = (colBase + lm) * LDS_B;
    FragU u;
    u.h[0] = *(const v8bf*)(sm + r + kb);        // K = kb .. kb+7
    u.h[1] = *(const v8bf*)(sm + r + kb + 8);    // K = kb+8 .. kb+15
    return u.v;
}

// ------------------------- weight convert + transpose -----------------------
// W[k][n] fp32  ->  WT[n][k] bf16   (so B-fragments read contiguous K)
__global__ void wconvT_kernel(const float* __restrict__ W, __bf16* __restrict__ WT,
                              int K, int N) {
    int idx = blockIdx.x * blockDim.x + threadIdx.x;
    if (idx >= K * N) return;
    int k = idx / N, n = idx - k * N;
    WT[(size_t)n * K + k] = (__bf16)W[idx];
}

// ------------------------------- RMSNorm ------------------------------------
__global__ __launch_bounds__(256) void rmsnorm_kernel(const float* __restrict__ x,
                                                      const float* __restrict__ w,
                                                      __bf16* __restrict__ out) {
    __shared__ float red[256];
    const int row = blockIdx.x;
    const int tid = threadIdx.x;
    const float4 v = ((const float4*)(x + (size_t)row * DM))[tid];
    float ss = v.x * v.x + v.y * v.y + v.z * v.z + v.w * v.w;
    red[tid] = ss;
    __syncthreads();
    for (int s = 128; s > 0; s >>= 1) {
        if (tid < s) red[tid] += red[tid + s];
        __syncthreads();
    }
    const float scale = rsqrtf(red[0] * (1.0f / DM) + EPSV);
    const int d = tid * 4;
    __bf16* o = out + (size_t)row * DM + d;
    o[0] = (__bf16)(v.x * scale * w[d + 0]);
    o[1] = (__bf16)(v.y * scale * w[d + 1]);
    o[2] = (__bf16)(v.z * scale * w[d + 2]);
    o[3] = (__bf16)(v.w * scale * w[d + 3]);
}

// ------------------- GEMM + bias + activation (sigmoid/tanh) ----------------
// C[M,N] = act(A[M,K] @ B[K,N] + bias), A bf16 row-major, BT bf16 [N,K].
template <int ACT>  // 0: sigmoid, 1: tanh
__global__ __launch_bounds__(256) void gemm_bias_act_kernel(
    const __bf16* __restrict__ A, const __bf16* __restrict__ BT,
    const float* __restrict__ bias, float* __restrict__ C,
    int M, int N, int K) {
    __shared__ __align__(16) __bf16 smA[2][BM * LDS_A];
    __shared__ __align__(16) __bf16 smB[2][BN * LDS_B];
    const int tid = threadIdx.x, lane = tid & 31, wave = tid >> 5;
    const int wm = wave >> 1, wn = wave & 1;      // 4x2 wave grid -> 32x32/wave
    const int mBase = blockIdx.y * BM, nBase = blockIdx.x * BN;

    const int aRow = tid >> 2, aCol = (tid & 3) * 8;   // A chunks (x2 rows apart 64)
    const int bRow = tid >> 2, bCol = (tid & 3) * 8;   // B^T chunk

    auto stage = [&](int buf, int k0) {
        async_copy16(&A[(size_t)(mBase + aRow) * K + k0 + aCol], &smA[buf][aRow * LDS_A + aCol]);
        async_copy16(&A[(size_t)(mBase + aRow + 64) * K + k0 + aCol], &smA[buf][(aRow + 64) * LDS_A + aCol]);
        async_copy16(&BT[(size_t)(nBase + bRow) * K + k0 + bCol], &smB[buf][bRow * LDS_B + bCol]);
    };

    v8f acc[2][2] = {};
    stage(0, 0);
    wait_async0();
    __syncthreads();

    int cur = 0;
    for (int k0 = 0; k0 < K; k0 += BK) {
        if (k0 + BK < K) {
            stage(cur ^ 1, k0 + BK);                        // overlap with WMMA
            if (k0 + 2 * BK < K)                            // L2 prefetch 2 ahead
                __builtin_prefetch(&A[(size_t)(mBase + aRow) * K + k0 + 2 * BK], 0, 0);
        }
        v16bf aF[2], bF[2];
#pragma unroll
        for (int mi = 0; mi < 2; ++mi) aF[mi] = load_frag_a(smA[cur], wm * 32 + mi * 16, lane);
#pragma unroll
        for (int ni = 0; ni < 2; ++ni) bF[ni] = load_frag_b(smB[cur], wn * 32 + ni * 16, lane);
#pragma unroll
        for (int mi = 0; mi < 2; ++mi)
#pragma unroll
            for (int ni = 0; ni < 2; ++ni)
                acc[mi][ni] = __builtin_amdgcn_wmma_f32_16x16x32_bf16(
                    false, aF[mi], false, bF[ni], (short)0, acc[mi][ni], false, false);
        wait_async0();
        __syncthreads();
        cur ^= 1;
    }

    const int lm = lane & 15, rOff = (lane & 16) ? 8 : 0;
#pragma unroll
    for (int mi = 0; mi < 2; ++mi)
#pragma unroll
        for (int ni = 0; ni < 2; ++ni) {
            const int col = nBase + wn * 32 + ni * 16 + lm;
            const float b = bias[col];
#pragma unroll
            for (int i = 0; i < 8; ++i) {
                const int row = mBase + wm * 32 + mi * 16 + i + rOff;
                float v = acc[mi][ni][i] + b;
                v = (ACT == 0) ? fsigmoid(v) : tanhf(v);
                C[(size_t)row * N + col] = v;
            }
        }
}

// -------------- fused SwiGLU up GEMM: T = silu(A@W1) * (A@W3) ---------------
__global__ __launch_bounds__(256) void gemm_swiglu_kernel(
    const __bf16* __restrict__ A, const __bf16* __restrict__ B1T,
    const __bf16* __restrict__ B2T, __bf16* __restrict__ T,
    int M, int N, int K) {
    __shared__ __align__(16) __bf16 smA[2][BM * LDS_A];
    __shared__ __align__(16) __bf16 smB1[2][BN * LDS_B];
    __shared__ __align__(16) __bf16 smB2[2][BN * LDS_B];
    const int tid = threadIdx.x, lane = tid & 31, wave = tid >> 5;
    const int wm = wave >> 1, wn = wave & 1;
    const int mBase = blockIdx.y * BM, nBase = blockIdx.x * BN;

    const int aRow = tid >> 2, aCol = (tid & 3) * 8;
    const int bRow = tid >> 2, bCol = (tid & 3) * 8;

    auto stage = [&](int buf, int k0) {
        async_copy16(&A[(size_t)(mBase + aRow) * K + k0 + aCol], &smA[buf][aRow * LDS_A + aCol]);
        async_copy16(&A[(size_t)(mBase + aRow + 64) * K + k0 + aCol], &smA[buf][(aRow + 64) * LDS_A + aCol]);
        async_copy16(&B1T[(size_t)(nBase + bRow) * K + k0 + bCol], &smB1[buf][bRow * LDS_B + bCol]);
        async_copy16(&B2T[(size_t)(nBase + bRow) * K + k0 + bCol], &smB2[buf][bRow * LDS_B + bCol]);
    };

    v8f acc1[2][2] = {}, acc2[2][2] = {};
    stage(0, 0);
    wait_async0();
    __syncthreads();

    int cur = 0;
    for (int k0 = 0; k0 < K; k0 += BK) {
        if (k0 + BK < K) {
            stage(cur ^ 1, k0 + BK);
            if (k0 + 2 * BK < K)
                __builtin_prefetch(&A[(size_t)(mBase + aRow) * K + k0 + 2 * BK], 0, 0);
        }
        v16bf aF[2], b1F[2], b2F[2];
#pragma unroll
        for (int mi = 0; mi < 2; ++mi) aF[mi] = load_frag_a(smA[cur], wm * 32 + mi * 16, lane);
#pragma unroll
        for (int ni = 0; ni < 2; ++ni) {
            b1F[ni] = load_frag_b(smB1[cur], wn * 32 + ni * 16, lane);
            b2F[ni] = load_frag_b(smB2[cur], wn * 32 + ni * 16, lane);
        }
#pragma unroll
        for (int mi = 0; mi < 2; ++mi)
#pragma unroll
            for (int ni = 0; ni < 2; ++ni) {
                acc1[mi][ni] = __builtin_amdgcn_wmma_f32_16x16x32_bf16(
                    false, aF[mi], false, b1F[ni], (short)0, acc1[mi][ni], false, false);
                acc2[mi][ni] = __builtin_amdgcn_wmma_f32_16x16x32_bf16(
                    false, aF[mi], false, b2F[ni], (short)0, acc2[mi][ni], false, false);
            }
        wait_async0();
        __syncthreads();
        cur ^= 1;
    }

    const int lm = lane & 15, rOff = (lane & 16) ? 8 : 0;
#pragma unroll
    for (int mi = 0; mi < 2; ++mi)
#pragma unroll
        for (int ni = 0; ni < 2; ++ni) {
            const int col = nBase + wn * 32 + ni * 16 + lm;
#pragma unroll
            for (int i = 0; i < 8; ++i) {
                const int row = mBase + wm * 32 + mi * 16 + i + rOff;
                const float u = acc1[mi][ni][i];
                const float v = acc2[mi][ni][i];
                T[(size_t)row * N + col] = (__bf16)(u * fsigmoid(u) * v);
            }
        }
}

// --------------- down GEMM + residual: out = res + A@W2 ---------------------
__global__ __launch_bounds__(256) void gemm_residual_kernel(
    const __bf16* __restrict__ A, const __bf16* __restrict__ BT,
    const float* __restrict__ res, float* __restrict__ out,
    int M, int N, int K) {
    __shared__ __align__(16) __bf16 smA[2][BM * LDS_A];
    __shared__ __align__(16) __bf16 smB[2][BN * LDS_B];
    const int tid = threadIdx.x, lane = tid & 31, wave = tid >> 5;
    const int wm = wave >> 1, wn = wave & 1;
    const int mBase = blockIdx.y * BM, nBase = blockIdx.x * BN;

    const int aRow = tid >> 2, aCol = (tid & 3) * 8;
    const int bRow = tid >> 2, bCol = (tid & 3) * 8;

    auto stage = [&](int buf, int k0) {
        async_copy16(&A[(size_t)(mBase + aRow) * K + k0 + aCol], &smA[buf][aRow * LDS_A + aCol]);
        async_copy16(&A[(size_t)(mBase + aRow + 64) * K + k0 + aCol], &smA[buf][(aRow + 64) * LDS_A + aCol]);
        async_copy16(&BT[(size_t)(nBase + bRow) * K + k0 + bCol], &smB[buf][bRow * LDS_B + bCol]);
    };

    v8f acc[2][2] = {};
    stage(0, 0);
    wait_async0();
    __syncthreads();

    int cur = 0;
    for (int k0 = 0; k0 < K; k0 += BK) {
        if (k0 + BK < K) {
            stage(cur ^ 1, k0 + BK);
            if (k0 + 2 * BK < K)
                __builtin_prefetch(&A[(size_t)(mBase + aRow) * K + k0 + 2 * BK], 0, 0);
        }
        v16bf aF[2], bF[2];
#pragma unroll
        for (int mi = 0; mi < 2; ++mi) aF[mi] = load_frag_a(smA[cur], wm * 32 + mi * 16, lane);
#pragma unroll
        for (int ni = 0; ni < 2; ++ni) bF[ni] = load_frag_b(smB[cur], wn * 32 + ni * 16, lane);
#pragma unroll
        for (int mi = 0; mi < 2; ++mi)
#pragma unroll
            for (int ni = 0; ni < 2; ++ni)
                acc[mi][ni] = __builtin_amdgcn_wmma_f32_16x16x32_bf16(
                    false, aF[mi], false, bF[ni], (short)0, acc[mi][ni], false, false);
        wait_async0();
        __syncthreads();
        cur ^= 1;
    }

    const int lm = lane & 15, rOff = (lane & 16) ? 8 : 0;
#pragma unroll
    for (int mi = 0; mi < 2; ++mi)
#pragma unroll
        for (int ni = 0; ni < 2; ++ni) {
            const int col = nBase + wn * 32 + ni * 16 + lm;
#pragma unroll
            for (int i = 0; i < 8; ++i) {
                const int row = mBase + wm * 32 + mi * 16 + i + rOff;
                const size_t o = (size_t)row * N + col;
                out[o] = res[o] + acc[mi][ni][i];
            }
        }
}

// -------------------- MinGRU chunked linear-recurrence scan -----------------
// h_t = g*h + (1-g)*c ; composition (A,B): h = A*h0 + B.
#define CHUNK 128
#define NCHUNK (LSEQ / CHUNK)   // 32

__global__ void scan_pass1_kernel(const float* __restrict__ g,
                                  const float* __restrict__ c,
                                  float* __restrict__ aggA, float* __restrict__ aggB) {
    int idx = blockIdx.x * blockDim.x + threadIdx.x;  // NCHUNK*NB*DM threads
    int d = idx & (DM - 1);
    int r = idx >> 10;
    int b = r & (NB - 1);
    int ch = r >> 2;
    float A = 1.0f, B = 0.0f;
    size_t base = ((size_t)b * LSEQ + (size_t)ch * CHUNK) * DM + d;
    for (int t = 0; t < CHUNK; ++t) {
        float gv = g[base], cv = c[base];
        A *= gv;
        B = gv * B + (1.0f - gv) * cv;
        base += DM;
    }
    size_t o = ((size_t)ch * NB + b) * DM + d;
    aggA[o] = A;
    aggB[o] = B;
}

__global__ void scan_pass2_kernel(const float* __restrict__ aggA,
                                  const float* __restrict__ aggB,
                                  float* __restrict__ carry) {
    int idx = blockIdx.x * blockDim.x + threadIdx.x;  // NB*DM threads
    int d = idx & (DM - 1);
    int b = idx >> 10;
    float h = 0.0f;
    for (int ch = 0; ch < NCHUNK; ++ch) {
        size_t o = ((size_t)ch * NB + b) * DM + d;
        carry[o] = h;
        h = aggA[o] * h + aggB[o];
    }
}

__global__ void scan_pass3_kernel(const float* __restrict__ g,
                                  const float* __restrict__ c,
                                  const float* __restrict__ carry,
                                  const float* __restrict__ x,
                                  float* __restrict__ xmid) {
    int idx = blockIdx.x * blockDim.x + threadIdx.x;
    int d = idx & (DM - 1);
    int r = idx >> 10;
    int b = r & (NB - 1);
    int ch = r >> 2;
    float h = carry[((size_t)ch * NB + b) * DM + d];
    size_t base = ((size_t)b * LSEQ + (size_t)ch * CHUNK) * DM + d;
    for (int t = 0; t < CHUNK; ++t) {
        float gv = g[base], cv = c[base];
        h = gv * h + (1.0f - gv) * cv;
        xmid[base] = x[base] + h;
        base += DM;
    }
}

// ---------------------------------------------------------------------------
extern "C" void kernel_launch(void* const* d_in, const int* in_sizes, int n_in,
                              void* d_out, int out_size, void* d_ws, size_t ws_size,
                              hipStream_t stream) {
    const float* x    = (const float*)d_in[0];
    const float* Wg   = (const float*)d_in[1];
    const float* bg   = (const float*)d_in[2];
    const float* Wc   = (const float*)d_in[3];
    const float* bc   = (const float*)d_in[4];
    const float* n1w  = (const float*)d_in[5];
    const float* n2w  = (const float*)d_in[6];
    const float* W1   = (const float*)d_in[7];
    const float* W3   = (const float*)d_in[8];
    const float* W2   = (const float*)d_in[9];
    float* out = (float*)d_out;

    char* ws = (char*)d_ws;
    size_t off = 0;
    auto alloc = [&](size_t bytes) -> void* {
        void* p = ws + off;
        off = (off + bytes + 255) & ~(size_t)255;
        return p;
    };
    __bf16* h_in  = (__bf16*)alloc((size_t)MTOK * DM * 2);
    __bf16* f_in  = (__bf16*)alloc((size_t)MTOK * DM * 2);
    __bf16* WgT   = (__bf16*)alloc((size_t)DM * DM * 2);
    __bf16* WcT   = (__bf16*)alloc((size_t)DM * DM * 2);
    __bf16* W1T   = (__bf16*)alloc((size_t)DM * DFF * 2);
    __bf16* W3T   = (__bf16*)alloc((size_t)DM * DFF * 2);
    __bf16* W2T   = (__bf16*)alloc((size_t)DFF * DM * 2);
    float*  gates = (float*)alloc((size_t)MTOK * DM * 4);
    float*  cands = (float*)alloc((size_t)MTOK * DM * 4);
    float*  xmid  = (float*)alloc((size_t)MTOK * DM * 4);
    __bf16* Tbuf  = (__bf16*)alloc((size_t)MTOK * DFF * 2);
    float*  aggA  = (float*)alloc((size_t)NCHUNK * NB * DM * 4);
    float*  aggB  = (float*)alloc((size_t)NCHUNK * NB * DM * 4);
    float*  carry = (float*)alloc((size_t)NCHUNK * NB * DM * 4);

    // 1) weights -> bf16, transposed (B^T layout for WMMA B-fragments)
    wconvT_kernel<<<(DM * DM + 255) / 256, 256, 0, stream>>>(Wg, WgT, DM, DM);
    wconvT_kernel<<<(DM * DM + 255) / 256, 256, 0, stream>>>(Wc, WcT, DM, DM);
    wconvT_kernel<<<(DM * DFF + 255) / 256, 256, 0, stream>>>(W1, W1T, DM, DFF);
    wconvT_kernel<<<(DM * DFF + 255) / 256, 256, 0, stream>>>(W3, W3T, DM, DFF);
    wconvT_kernel<<<(DFF * DM + 255) / 256, 256, 0, stream>>>(W2, W2T, DFF, DM);

    // 2) h_in = rmsnorm(x) (bf16)
    rmsnorm_kernel<<<MTOK, 256, 0, stream>>>(x, n1w, h_in);

    // 3) gates = sigmoid(h_in@Wg + bg), cands = tanh(h_in@Wc + bc)
    dim3 gridG(DM / BN, MTOK / BM);
    gemm_bias_act_kernel<0><<<gridG, 256, 0, stream>>>(h_in, WgT, bg, gates, MTOK, DM, DM);
    gemm_bias_act_kernel<1><<<gridG, 256, 0, stream>>>(h_in, WcT, bc, cands, MTOK, DM, DM);

    // 4) chunked linear scan; xmid = x + h
    scan_pass1_kernel<<<(NCHUNK * NB * DM) / 256, 256, 0, stream>>>(gates, cands, aggA, aggB);
    scan_pass2_kernel<<<(NB * DM) / 256, 256, 0, stream>>>(aggA, aggB, carry);
    scan_pass3_kernel<<<(NCHUNK * NB * DM) / 256, 256, 0, stream>>>(gates, cands, carry, x, xmid);

    // 5) f_in = rmsnorm(xmid)
    rmsnorm_kernel<<<MTOK, 256, 0, stream>>>(xmid, n2w, f_in);

    // 6) T = silu(f_in@W1) * (f_in@W3)   (fused dual GEMM, bf16 out)
    dim3 gridU(DFF / BN, MTOK / BM);
    gemm_swiglu_kernel<<<gridU, 256, 0, stream>>>(f_in, W1T, W3T, Tbuf, MTOK, DFF, DM);

    // 7) out = xmid + T@W2
    dim3 gridD(DM / BN, MTOK / BM);
    gemm_residual_kernel<<<gridD, 256, 0, stream>>>(Tbuf, W2T, xmid, out, MTOK, DM, DFF);
}
